// MultiHeadHyperNet_15522011808532
// MI455X (gfx1250) — matrix operand order repaired and checked by reading, not compiled, not run on hardware
//
#include <hip/hip_runtime.h>
#include <hip/hip_bf16.h>

typedef __attribute__((ext_vector_type(16))) _Float16 v16h;
typedef __attribute__((ext_vector_type(8)))  _Float16 v8h;
typedef __attribute__((ext_vector_type(8)))  float    v8f;

#define DD    256
#define CC    10
#define TT    15
#define HH    5
#define II    15
#define LL    16
#define ENC   64
#define PP    60240      // T*I*(D+1) + T*L*C + T
#define NS    2048
#define NQ    32768
#define NTREE 75         // H*T
#define SW    57600      // T*I*D
#define SB    225        // T*I
#define LF    2400       // T*L*C

// ---- workspace layout (bytes) ----
// zero span covers W16 + bias + ctxsum
#define OFF_W16     0u                       // 75*16*256*2 = 614400
#define OFF_BIAS    614400u                  // 75*16*4    = 4800
#define OFF_CTXSUM  619264u                  // 320*4      = 1280
#define ZERO_BYTES  620544u
#define OFF_LEAFRAW 620544u                  // 48000
#define OFF_LEAFP   668672u                  // 48000
#define OFF_TREERAW 716800u                  // 300
#define OFF_WCOMB   717056u                  // 300
#define OFF_HEADW   717568u                  // 20
#define OFF_PBUF    717824u                  // 2560
#define OFF_XQ16    720384u                  // NQ*D*2 = 16777216

__device__ __forceinline__ float gelu_exact(float x) {
    return 0.5f * x * (1.0f + erff(x * 0.70710678118654752f));
}

// -------------------- zero init --------------------
__global__ void k_zero(unsigned* __restrict__ p, int n) {
    int i = blockIdx.x * blockDim.x + threadIdx.x;
    if (i < n) p[i] = 0u;
}

// -------------------- X_query f32 -> f16 --------------------
__global__ void k_cvtq(const float* __restrict__ xq, _Float16* __restrict__ xq16) {
    int i = blockIdx.x * blockDim.x + threadIdx.x;   // one float4 per thread
    float4 v = ((const float4*)xq)[i];
    _Float16* o = xq16 + (size_t)i * 4;
    o[0] = (_Float16)v.x; o[1] = (_Float16)v.y;
    o[2] = (_Float16)v.z; o[3] = (_Float16)v.w;
}

// -------------------- support encoder: one block = one (head, row) --------------------
__global__ void k_encode(const float* __restrict__ Xs,
                         const float* __restrict__ w1, const float* __restrict__ b1,
                         const float* __restrict__ g1, const float* __restrict__ bt1,
                         const float* __restrict__ w2, const float* __restrict__ b2,
                         const float* __restrict__ g2, const float* __restrict__ bt2,
                         float* __restrict__ ctxsum) {
    __shared__ float xs[DD];
    __shared__ float red[ENC];
    __shared__ float h1s[ENC];
    const int e  = threadIdx.x;           // 0..63
    const int bh = blockIdx.x / NS;
    const int n  = blockIdx.x % NS;

    float4 x4 = ((const float4*)Xs)[(size_t)n * (DD / 4) + e];
    xs[e * 4 + 0] = x4.x; xs[e * 4 + 1] = x4.y; xs[e * 4 + 2] = x4.z; xs[e * 4 + 3] = x4.w;
    __syncthreads();

    float acc = b1[bh * ENC + e];
    for (int d = 0; d < DD; ++d)
        acc = fmaf(xs[d], w1[((size_t)(bh * DD + d)) * ENC + e], acc);

    red[e] = acc; __syncthreads();
    float s1 = 0.f, s2 = 0.f;
    for (int i = 0; i < ENC; ++i) { float t = red[i]; s1 += t; s2 += t * t; }
    __syncthreads();
    float m = s1 * (1.0f / ENC), var = s2 * (1.0f / ENC) - m * m;
    float nv = (acc - m) * rsqrtf(var + 1e-5f) * g1[bh * ENC + e] + bt1[bh * ENC + e];
    h1s[e] = gelu_exact(nv);
    __syncthreads();

    float acc2 = b2[bh * ENC + e];
    for (int q = 0; q < ENC; ++q)
        acc2 = fmaf(h1s[q], w2[((size_t)(bh * ENC + q)) * ENC + e], acc2);

    red[e] = acc2; __syncthreads();
    s1 = 0.f; s2 = 0.f;
    for (int i = 0; i < ENC; ++i) { float t = red[i]; s1 += t; s2 += t * t; }
    __syncthreads();
    m = s1 * (1.0f / ENC); var = s2 * (1.0f / ENC) - m * m;
    float nv2 = (acc2 - m) * rsqrtf(var + 1e-5f) * g2[bh * ENC + e] + bt2[bh * ENC + e];
    atomicAdd(&ctxsum[bh * ENC + e], gelu_exact(nv2));
}

// -------------------- ctx -> p (H,128), plus head softmax --------------------
__global__ void k_ctx2p(const float* __restrict__ ctxsum,
                        const float* __restrict__ pw1, const float* __restrict__ pb1,
                        const float* __restrict__ plg, const float* __restrict__ plb,
                        const float* __restrict__ hwraw,
                        float* __restrict__ pG, float* __restrict__ headwG) {
    __shared__ float ctxs[ENC];
    __shared__ float red[128];
    const int f = threadIdx.x;            // 0..127
    for (int h = 0; h < HH; ++h) {
        if (f < ENC) ctxs[f] = ctxsum[h * ENC + f] * (1.0f / NS);
        __syncthreads();
        float t1 = pb1[h * 128 + f];
        for (int e = 0; e < ENC; ++e)
            t1 = fmaf(ctxs[e], pw1[((size_t)(h * ENC + e)) * 128 + f], t1);
        red[f] = t1; __syncthreads();
        float s1 = 0.f, s2 = 0.f;
        for (int i = 0; i < 128; ++i) { float t = red[i]; s1 += t; s2 += t * t; }
        __syncthreads();
        float m = s1 * (1.0f / 128.0f), var = s2 * (1.0f / 128.0f) - m * m;
        float nv = (t1 - m) * rsqrtf(var + 1e-5f) * plg[h * 128 + f] + plb[h * 128 + f];
        pG[h * 128 + f] = gelu_exact(nv);
        __syncthreads();
    }
    if (f == 0) {
        float mx = -1e30f;
        for (int h = 0; h < HH; ++h) mx = fmaxf(mx, hwraw[h]);
        float s = 0.f, e[HH];
        for (int h = 0; h < HH; ++h) { e[h] = __expf(hwraw[h] - mx); s += e[h]; }
        for (int h = 0; h < HH; ++h) headwG[h] = e[h] / s;
    }
}

// -------------------- param generation (memory-bound stream of pg_w2) --------------------
__global__ void k_params(const float* __restrict__ pG,
                         const float* __restrict__ pw2, const float* __restrict__ pb2,
                         _Float16* __restrict__ W16, float* __restrict__ biasG,
                         float* __restrict__ leafrawG, float* __restrict__ treerawG) {
    __shared__ float pl[256];             // rows h0 and h0+1 of p (clamped)
    const int tid = threadIdx.x;
    long long gid = (long long)blockIdx.x * blockDim.x + tid;
    const int h0 = (int)(((long long)blockIdx.x * blockDim.x) / PP);
    {
        int idx = h0 * 128 + tid;
        int mxi = HH * 128 - 1;
        pl[tid] = pG[idx < mxi ? idx : mxi];
    }
    __syncthreads();
    if (gid >= (long long)HH * PP) return;
    int h = (int)(gid / PP);
    int j = (int)(gid % PP);
    const float* prow = &pl[(h - h0) * 128];
    float s = pb2[(size_t)h * PP + j];
    for (int f = 0; f < 128; ++f)
        s = fmaf(prow[f], pw2[((size_t)(h * 128 + f)) * PP + j], s);
    float val = tanhf(s) * 2.0f;

    if (j < SW) {
        int t = j / (II * DD), r = j % (II * DD);
        int i = r / DD, d = r % DD;
        int tree = h * TT + t;
        W16[(size_t)tree * 4096 + i * DD + d] = (_Float16)val;   // column-major per tree
    } else if (j < SW + SB) {
        int r = j - SW;
        int t = r / II, i = r % II;
        biasG[(h * TT + t) * 16 + i] = val;
    } else if (j < SW + SB + LF) {
        int r = j - SW - SB;
        int t = r / (LL * CC), q = r % (LL * CC);
        leafrawG[(h * TT + t) * (LL * CC) + q] = val;
    } else {
        int t = j - SW - SB - LF;
        treerawG[h * TT + t] = val;
    }
}

// -------------------- leaf softmax(logits/temp) --------------------
__global__ void k_leafsm(const float* __restrict__ leafrawG, const float* __restrict__ tempP,
                         float* __restrict__ leafpG) {
    int row = blockIdx.x * blockDim.x + threadIdx.x;   // (tree, leaf)
    if (row >= NTREE * LL) return;
    float temp = fminf(fmaxf(tempP[0], 0.1f), 2.0f);
    float invt = 1.0f / temp;
    const float* src = leafrawG + (size_t)row * CC;
    float v[CC], mx = -1e30f;
    for (int c = 0; c < CC; ++c) { v[c] = src[c] * invt; mx = fmaxf(mx, v[c]); }
    float s = 0.f;
    for (int c = 0; c < CC; ++c) { v[c] = __expf(v[c] - mx); s += v[c]; }
    float inv = 1.0f / s;
    float* dst = leafpG + (size_t)row * CC;
    for (int c = 0; c < CC; ++c) dst[c] = v[c] * inv;
}

// -------------------- tree weights: softmax over T, fold in head weight --------------------
__global__ void k_treew(const float* __restrict__ treerawG, const float* __restrict__ headwG,
                        float* __restrict__ wcombG) {
    int tid = threadIdx.x;
    if (tid >= NTREE) return;
    int h = tid / TT, t = tid % TT;
    const float* src = treerawG + h * TT;
    float mx = -1e30f;
    for (int k = 0; k < TT; ++k) mx = fmaxf(mx, src[k]);
    float s = 0.f;
    for (int k = 0; k < TT; ++k) s += __expf(src[k] - mx);
    wcombG[tid] = (__expf(src[t] - mx) / s) * headwG[h];
}

// -------------------- main query kernel: WMMA GEMM + soft tree --------------------
__launch_bounds__(128, 1)
__global__ void k_main(const _Float16* __restrict__ Xq16, const _Float16* __restrict__ W16,
                       const float* __restrict__ biasG, const float* __restrict__ leafpG,
                       const float* __restrict__ wcombG, const float* __restrict__ tempP,
                       float* __restrict__ out) {
    __shared__ float leafAll[NTREE * LL * CC];   // 48000 B: all leaf probabilities
    __shared__ float biasAll[NTREE * 16];        // 4800 B
    __shared__ float wAll[NTREE];                // 300 B
    __shared__ float decLDS[4][32 * 17];         // per-wave, stride 17 => conflict-free

    const int tid  = threadIdx.x;
    const int wave = tid >> 5;
    const int lane = tid & 31;
    const int lrow = lane & 15;            // N column / A row within tile
    const int lhi  = lane >> 4;            // half-wave select
    const int rowBase = blockIdx.x * 128 + wave * 32;

    // one-time staging of all per-tree constants
    for (int s = tid; s < NTREE * LL * CC; s += 128) leafAll[s] = leafpG[s];
    for (int s = tid; s < NTREE * 16; s += 128)      biasAll[s] = biasG[s];
    for (int s = tid; s < NTREE; s += 128)           wAll[s] = wcombG[s];

    float temp = fminf(fmaxf(tempP[0], 0.1f), 2.0f);
    float invt = 1.0f / temp;

    // A tiles resident in registers: 2 row-tiles x 8 K-chunks (K=256)
    v16h a[2][8];
#pragma unroll
    for (int t2 = 0; t2 < 2; ++t2) {
        const _Float16* rp = Xq16 + (size_t)(rowBase + t2 * 16 + lrow) * DD;
#pragma unroll
        for (int k = 0; k < 8; ++k) {
            int kb = k * 32 + lhi * 8;
            v8h lo = *(const v8h*)(rp + kb);
            v8h hi = *(const v8h*)(rp + kb + 16);
            a[t2][k] = __builtin_shufflevector(lo, hi, 0, 1, 2, 3, 4, 5, 6, 7,
                                               8, 9, 10, 11, 12, 13, 14, 15);
        }
    }

    float acc[CC];
#pragma unroll
    for (int c = 0; c < CC; ++c) acc[c] = 0.f;

    __syncthreads();   // staging complete; loop below is barrier-free

    const _Float16* wp = W16 + lrow * DD + lhi * 8;
    float* dl = &decLDS[wave][0];

    for (int tree = 0; tree < NTREE; ++tree) {
        // preload the whole tree's B (16 x b128) so WMMA waits pipeline down
        v16h bb[8];
#pragma unroll
        for (int k = 0; k < 8; ++k) {
            const _Float16* bp = wp + (size_t)tree * 4096 + k * 32;
            v8h blo = *(const v8h*)(bp);
            v8h bhi = *(const v8h*)(bp + 16);
            bb[k] = __builtin_shufflevector(blo, bhi, 0, 1, 2, 3, 4, 5, 6, 7,
                                            8, 9, 10, 11, 12, 13, 14, 15);
        }

        v8f c0 = {}, c1 = {};
#pragma unroll
        for (int k = 0; k < 8; ++k) {
            c0 = __builtin_amdgcn_wmma_f32_16x16x32_f16(false, a[0][k], false, bb[k],
                                                        (short)0, c0, false, false);
            c1 = __builtin_amdgcn_wmma_f32_16x16x32_f16(false, a[1][k], false, bb[k],
                                                        (short)0, c1, false, false);
        }

        // sigmoid((dot + b)/temp), scatter decisions to per-wave LDS
        float bi = biasAll[tree * 16 + lrow];
#pragma unroll
        for (int v = 0; v < 8; ++v) {
            int r0 = v + lhi * 8;
            float x0 = (c0[v] + bi) * invt;
            dl[r0 * 17 + lrow] = 1.0f / (1.0f + __expf(-x0));
            float x1 = (c1[v] + bi) * invt;
            dl[(16 + r0) * 17 + lrow] = 1.0f / (1.0f + __expf(-x1));
        }
        __builtin_amdgcn_wave_barrier();   // wave-lockstep LDS exchange (no block barrier)

        // one query row per lane: walk the soft tree
        float dv[II];
#pragma unroll
        for (int i = 0; i < II; ++i) dv[i] = dl[lane * 17 + i];

        float reach[LL];
        reach[0] = 1.f;
#pragma unroll
        for (int dep = 0; dep < 4; ++dep) {
            int start = (1 << dep) - 1, n = 1 << dep;
            float nr[LL];
#pragma unroll
            for (int kk = 0; kk < n; ++kk) {
                float dd = dv[start + kk];
                nr[2 * kk]     = reach[kk] * (1.f - dd);
                nr[2 * kk + 1] = reach[kk] * dd;
            }
#pragma unroll
            for (int kk = 0; kk < 2 * n; ++kk) reach[kk] = nr[kk];
        }

        const float* lp = &leafAll[tree * (LL * CC)];
        float pr[CC];
#pragma unroll
        for (int c = 0; c < CC; ++c) pr[c] = 0.f;
#pragma unroll
        for (int l = 0; l < LL; ++l) {
            float rl = reach[l];
#pragma unroll
            for (int c = 0; c < CC; ++c) pr[c] = fmaf(rl, lp[l * CC + c], pr[c]);
        }
        float w = wAll[tree];
#pragma unroll
        for (int c = 0; c < CC; ++c) acc[c] = fmaf(w, pr[c], acc[c]);
        __builtin_amdgcn_wave_barrier();   // keep dec reads before next tree's writes
    }

    int n = rowBase + lane;
#pragma unroll
    for (int c = 0; c < CC; ++c) out[(size_t)n * CC + c] = acc[c];
}

extern "C" void kernel_launch(void* const* d_in, const int* in_sizes, int n_in,
                              void* d_out, int out_size, void* d_ws, size_t ws_size,
                              hipStream_t stream) {
    const float* Xs    = (const float*)d_in[0];
    const float* Xq    = (const float*)d_in[1];
    const float* ew1   = (const float*)d_in[2];
    const float* eb1   = (const float*)d_in[3];
    const float* l1g   = (const float*)d_in[4];
    const float* l1b   = (const float*)d_in[5];
    const float* ew2   = (const float*)d_in[6];
    const float* eb2   = (const float*)d_in[7];
    const float* l2g   = (const float*)d_in[8];
    const float* l2b   = (const float*)d_in[9];
    const float* pw1   = (const float*)d_in[10];
    const float* pb1   = (const float*)d_in[11];
    const float* plg   = (const float*)d_in[12];
    const float* plb   = (const float*)d_in[13];
    const float* pw2   = (const float*)d_in[14];
    const float* pb2   = (const float*)d_in[15];
    const float* hwraw = (const float*)d_in[16];
    const float* tempP = (const float*)d_in[17];
    float* out = (float*)d_out;

    char* w = (char*)d_ws;
    _Float16* W16      = (_Float16*)(w + OFF_W16);
    float*    biasG    = (float*)(w + OFF_BIAS);
    float*    ctxsum   = (float*)(w + OFF_CTXSUM);
    float*    leafrawG = (float*)(w + OFF_LEAFRAW);
    float*    leafpG   = (float*)(w + OFF_LEAFP);
    float*    treerawG = (float*)(w + OFF_TREERAW);
    float*    wcombG   = (float*)(w + OFF_WCOMB);
    float*    headwG   = (float*)(w + OFF_HEADW);
    float*    pG       = (float*)(w + OFF_PBUF);
    _Float16* Xq16     = (_Float16*)(w + OFF_XQ16);

    const int zeroN = ZERO_BYTES / 4;
    k_zero<<<(zeroN + 255) / 256, 256, 0, stream>>>((unsigned*)w, zeroN);
    k_cvtq<<<(NQ * DD / 4) / 256, 256, 0, stream>>>(Xq, Xq16);
    k_encode<<<HH * NS, ENC, 0, stream>>>(Xs, ew1, eb1, l1g, l1b, ew2, eb2, l2g, l2b, ctxsum);
    k_ctx2p<<<1, 128, 0, stream>>>(ctxsum, pw1, pb1, plg, plb, hwraw, pG, headwG);
    k_params<<<(HH * PP + 255) / 256, 256, 0, stream>>>(pG, pw2, pb2, W16, biasG, leafrawG, treerawG);
    k_leafsm<<<(NTREE * LL + 127) / 128, 128, 0, stream>>>(leafrawG, tempP, leafpG);
    k_treew<<<1, 128, 0, stream>>>(treerawG, headwG, wcombG);
    k_main<<<NQ / 128, 128, 0, stream>>>(Xq16, W16, biasG, leafpG, wcombG, tempP, out);
}
